// GAOTEncoder_36258113912891
// MI455X (gfx1250) — matrix operand order repaired and checked by reading, not compiled, not run on hardware
//
#include <hip/hip_runtime.h>
#include <math.h>

// ---- CDNA5 WMMA / TDM vector types ----
typedef __attribute__((ext_vector_type(16))) _Float16 v16h;
typedef __attribute__((ext_vector_type(8)))  _Float16 v8h;
typedef __attribute__((ext_vector_type(4)))  _Float16 v4h;
typedef __attribute__((ext_vector_type(2)))  _Float16 v2h;
typedef __attribute__((ext_vector_type(8)))  float    v8f;
typedef __attribute__((ext_vector_type(4)))  unsigned int u32x4;
typedef __attribute__((ext_vector_type(8)))  int          i32x8;
typedef __attribute__((ext_vector_type(4)))  int          i32x4;

#define NN_H   128
#define NN_A   64
#define NN_L   3
#define NN_S   2
#define QKVW   512          // fused per-layer QKV output width: 2 scales * (64+64+128)
#define AGGW   256          // concat of 2 scale aggregations (2*128)

__device__ __forceinline__ float gelu_tanh(float x) {
    float x3 = x * x * x;
    return 0.5f * x * (1.0f + tanhf(0.7978845608028654f * (x + 0.044715f * x3)));
}

// ---------------------------------------------------------------------------
// TDM 1-D tile load: copy nelem fp16 elements (contiguous) global -> LDS.
// D# per CDNA5 ISA ch.8: group0 {count=1, lds_addr, global_addr[56:0], type=2},
// group1 {data_size=2B, tensor_dim0=tile_dim0=nelem, tensor_dim1=tile_dim1=1,
//         tensor_dim0_stride=nelem}. Groups 2/3 zero (<=2D tensor).
// Tracked by TENSORcnt. Issued by one wave per block; EXEC is ignored by TDM.
// This toolchain uses the 6-arg builtin form (clang-23 / therock-10.0).
// ---------------------------------------------------------------------------
__device__ __forceinline__ void tdm_load_1d_f16(unsigned lds_byte_off,
                                                const void* gptr, unsigned nelem) {
    unsigned long long ga = (unsigned long long)(uintptr_t)gptr;
    u32x4 g0;
    g0[0] = 1u;                                          // count=1, user descriptor
    g0[1] = lds_byte_off;                                // lds_addr
    g0[2] = (unsigned)(ga & 0xFFFFFFFFu);                // global_addr[31:0]
    g0[3] = (unsigned)((ga >> 32) & 0x01FFFFFFu)         // global_addr[56:32]
          | (2u << 30);                                  // type = 2 ("image")
    i32x8 g1;
    g1[0] = 0x00010000;                                  // data_size=1 (2 bytes)
    g1[1] = (int)((nelem & 0xFFFFu) << 16);              // tensor_dim0[15:0]
    g1[2] = (int)(((nelem >> 16) & 0xFFFFu)              // tensor_dim0[31:16]
          | (1u << 16));                                 // tensor_dim1 = 1
    g1[3] = (int)((nelem & 0xFFFFu) << 16);              // tile_dim0
    g1[4] = 1;                                           // tile_dim1 = 1
    g1[5] = (int)nelem;                                  // tensor_dim0_stride[31:0]
    g1[6] = 0;
    g1[7] = 0;
    i32x4 z4 = {0, 0, 0, 0};
    i32x8 z8 = {0, 0, 0, 0, 0, 0, 0, 0};
    __builtin_amdgcn_tensor_load_to_lds(g0, g1, z4, z4, z8, 0);
}

// ---------------------------------------------------------------------------
// 1) Geometry embedding: per node, per scale: mean rel offset (2) + mean dist
// ---------------------------------------------------------------------------
__global__ void geom_kernel(const float* __restrict__ coords,
                            const int* __restrict__ ip0, const int* __restrict__ ix0,
                            const int* __restrict__ ip1, const int* __restrict__ ix1,
                            float* __restrict__ geom, int N) {
    int node = blockIdx.x * blockDim.x + threadIdx.x;
    if (node >= N) return;
    float cx = coords[node * 2 + 0], cy = coords[node * 2 + 1];
    const int* ips[2] = {ip0, ip1};
    const int* ixs[2] = {ix0, ix1};
#pragma unroll
    for (int s = 0; s < 2; ++s) {
        int b = ips[s][node], e = ips[s][node + 1];
        float sx = 0.f, sy = 0.f, sd = 0.f;
        for (int t = b; t < e; ++t) {
            int c = ixs[s][t];
            float rx = coords[c * 2 + 0] - cx;
            float ry = coords[c * 2 + 1] - cy;
            sx += rx; sy += ry; sd += sqrtf(rx * rx + ry * ry);
        }
        float inv = 1.0f / fmaxf((float)(e - b), 1.0f);
        geom[node * 6 + s * 3 + 0] = sx * inv;
        geom[node * 6 + s * 3 + 1] = sy * inv;
        geom[node * 6 + s * 3 + 2] = sd * inv;
    }
}

// ---------------------------------------------------------------------------
// 2) Pack weights to fp16, column-major (K contiguous per output column).
//    wqkv[l][col][k], col = s*256 + {0..63:Q, 64..127:K, 128..255:V}
//    wot [l][col][k], col in [0,128), k in [0,256)
// ---------------------------------------------------------------------------
__global__ void pack_weights(const float* __restrict__ Wq, const float* __restrict__ Wk,
                             const float* __restrict__ Wv, const float* __restrict__ Wo,
                             _Float16* __restrict__ wqkv, _Float16* __restrict__ wot) {
    const int total1 = NN_L * QKVW * NN_H;   // 3*512*128
    const int total2 = NN_L * NN_H * AGGW;   // 3*128*256
    for (int i = blockIdx.x * blockDim.x + threadIdx.x; i < total1 + total2;
         i += gridDim.x * blockDim.x) {
        if (i < total1) {
            int l = i / (QKVW * NN_H);
            int rem = i % (QKVW * NN_H);
            int col = rem / NN_H;
            int k   = rem % NN_H;
            int s = col / 256, r = col % 256;
            float v;
            if (r < 64)        v = Wq[(((l * 2 + s) * NN_H + k) * NN_A) + r];
            else if (r < 128)  v = Wk[(((l * 2 + s) * NN_H + k) * NN_A) + (r - 64)];
            else               v = Wv[(((l * 2 + s) * NN_H + k) * NN_H) + (r - 128)];
            wqkv[i] = (_Float16)v;
        } else {
            int j = i - total1;
            int l = j / (NN_H * AGGW);
            int rem = j % (NN_H * AGGW);
            int col = rem / AGGW;
            int k   = rem % AGGW;
            wot[j] = (_Float16)Wo[((l * AGGW + k) * NN_H) + col];
        }
    }
}

// ---------------------------------------------------------------------------
// 3) Prologue: x = gelu([fields(16)|geom(6)] @ W_in(22x128) + b_in)
// ---------------------------------------------------------------------------
__global__ void prologue_kernel(const float* __restrict__ fields, const float* __restrict__ geom,
                                const float* __restrict__ W_in, const float* __restrict__ b_in,
                                float* __restrict__ x32, _Float16* __restrict__ x16, int N) {
    __shared__ float sin_[22];
    int node = blockIdx.x;
    int j = threadIdx.x;   // 128 threads
    if (node >= N) return;
    if (j < 16)      sin_[j] = fields[node * 16 + j];
    else if (j < 22) sin_[j] = geom[node * 6 + (j - 16)];
    __syncthreads();
    float acc = b_in[j];
#pragma unroll
    for (int i = 0; i < 22; ++i) acc += sin_[i] * W_in[i * NN_H + j];
    float y = gelu_tanh(acc);
    x32[(size_t)node * NN_H + j] = y;
    x16[(size_t)node * NN_H + j] = (_Float16)y;
}

// ---------------------------------------------------------------------------
// 4) WMMA QKV GEMM: Out[N,512] = X16[N,128] @ Wqkv^T  (W col-major [512][128])
//    Block: 8 waves x 16-row tiles, 64 shared columns. The 16KB weight tile is
//    staged into LDS once per block by the Tensor Data Mover; B-fragments are
//    then ds_load'ed. One wave -> 16x64 tile: 4 accs, 4 K-steps of
//    v_wmma_f32_16x16x32_f16.
// ---------------------------------------------------------------------------
__global__ __launch_bounds__(256) void wmma_gemm_qkv(const _Float16* __restrict__ X,
                                                     const _Float16* __restrict__ W,
                                                     _Float16* __restrict__ Out, int N) {
    __shared__ _Float16 shB[64 * NN_H];     // 16 KB: 64 cols x 128 K, col-major
    int lane = threadIdx.x & 31;
    int wave = threadIdx.x >> 5;
    int rowBase = (blockIdx.x * 8 + wave) * 16;
    int colBase = blockIdx.y * 64;
    int half = lane >> 4;
    int l16  = lane & 15;

    if (threadIdx.x < 32) {                 // wave 0 issues the TDM copy
        tdm_load_1d_f16((unsigned)(uintptr_t)shB,
                        W + (size_t)colBase * NN_H, 64 * NN_H);
        __builtin_amdgcn_s_wait_tensorcnt(0);
    }
    __syncthreads();
    if (rowBase >= N) return;               // wave-uniform (after the barrier)

    v8f acc[4] = {};
    // A k order (this kernel's contiguous choice, same as B): lanes<16 hold K 0..15, lanes>=16 hold K 16..31
    const _Float16* xrow = X + (size_t)(rowBase + l16) * NN_H + half * 16;
#pragma unroll
    for (int kc = 0; kc < NN_H; kc += 32) {
        v8h a0 = *(const v8h*)(xrow + kc);
        v8h a1 = *(const v8h*)(xrow + kc + 8);
        v16h A;
#pragma unroll
        for (int i = 0; i < 8; ++i) { A[i] = a0[i]; A[i + 8] = a1[i]; }
#pragma unroll
        for (int t = 0; t < 4; ++t) {
            // B k order (this kernel's contiguous choice, same as A): lanes<16 hold K 0..15, lanes>=16 hold K 16..31
            const _Float16* bp = shB + (t * 16 + l16) * NN_H + kc + half * 16;
            v16h B = *(const v16h*)bp;
            acc[t] = __builtin_amdgcn_wmma_f32_16x16x32_f16(false, A, false, B,
                                                            (short)0, acc[t], false, false);
        }
    }
    // C layout: VGPR g -> M = g + half*8, N = lane&15
#pragma unroll
    for (int t = 0; t < 4; ++t)
#pragma unroll
        for (int g = 0; g < 8; ++g) {
            int row = rowBase + g + half * 8;
            int col = colBase + t * 16 + l16;
            Out[(size_t)row * QKVW + col] = (_Float16)acc[t][g];
        }
}

// ---------------------------------------------------------------------------
// 5) Normalize q and k rows: one wave per (node, scale, q|k); 64 dims = 2/lane
// ---------------------------------------------------------------------------
__global__ __launch_bounds__(256) void normalize_qk(_Float16* __restrict__ qkv, int N) {
    int lane = threadIdx.x & 31;
    int wid = (blockIdx.x * blockDim.x + threadIdx.x) >> 5;
    if (wid >= N * 4) return;
    int node = wid >> 2;
    int s = (wid >> 1) & 1;
    int which = wid & 1;
    _Float16* p = qkv + (size_t)node * QKVW + s * 256 + which * 64 + lane * 2;
    v2h v = *(v2h*)p;
    float a = (float)v[0], b = (float)v[1];
    float ss = a * a + b * b;
#pragma unroll
    for (int m = 1; m < 32; m <<= 1) ss += __shfl_xor(ss, m, 32);
    float sc = 1.0f / (sqrtf(ss) + 1e-6f);
    v[0] = (_Float16)(a * sc);
    v[1] = (_Float16)(b * sc);
    *(v2h*)p = v;
}

// ---------------------------------------------------------------------------
// 6) Edge attention (one wave per node, online softmax, 128-dim V = 4/lane)
// ---------------------------------------------------------------------------
__global__ __launch_bounds__(256) void edge_attn(const _Float16* __restrict__ qkv,
                                                 const int* __restrict__ indptr,
                                                 const int* __restrict__ indices,
                                                 _Float16* __restrict__ agg, int N,
                                                 int qbase, int kbase, int vbase, int aggbase) {
    int lane = threadIdx.x & 31;
    int wave = threadIdx.x >> 5;
    int node = blockIdx.x * 8 + wave;
    if (node >= N) return;
    v2h qv = *(const v2h*)(qkv + (size_t)node * QKVW + qbase + lane * 2);
    float q0 = (float)qv[0], q1 = (float)qv[1];
    int b = indptr[node], e = indptr[node + 1];
    float m = -3.0e38f, d = 0.f;
    float a0 = 0.f, a1 = 0.f, a2 = 0.f, a3 = 0.f;
    for (int t = b; t < e; ++t) {
        int c = indices[t];
        v2h kv = *(const v2h*)(qkv + (size_t)c * QKVW + kbase + lane * 2);
        float sc = q0 * (float)kv[0] + q1 * (float)kv[1];
#pragma unroll
        for (int mm = 1; mm < 32; mm <<= 1) sc += __shfl_xor(sc, mm, 32);
        float mn = fmaxf(m, sc);
        float corr = __expf(m - mn);     // first iter: exp(-inf) -> 0, zeroes stale state
        float p = __expf(sc - mn);
        d = d * corr + p;
        v4h vv = *(const v4h*)(qkv + (size_t)c * QKVW + vbase + lane * 4);
        a0 = a0 * corr + p * (float)vv[0];
        a1 = a1 * corr + p * (float)vv[1];
        a2 = a2 * corr + p * (float)vv[2];
        a3 = a3 * corr + p * (float)vv[3];
        m = mn;
    }
    float inv = d > 0.f ? 1.0f / d : 0.f;
    v4h out;
    out[0] = (_Float16)(a0 * inv);
    out[1] = (_Float16)(a1 * inv);
    out[2] = (_Float16)(a2 * inv);
    out[3] = (_Float16)(a3 * inv);
    *(v4h*)(agg + (size_t)node * AGGW + aggbase + lane * 4) = out;
}

// ---------------------------------------------------------------------------
// 7) WMMA output GEMM + bias + GELU + residual (in place on x32)
//    h = Agg[N,256] @ Wo ; x = x + gelu(h + bo). 32KB weight tile staged via TDM.
// ---------------------------------------------------------------------------
__global__ __launch_bounds__(256) void wmma_gemm_out(const _Float16* __restrict__ Agg,
                                                     const _Float16* __restrict__ Wt,
                                                     const float* __restrict__ bo,
                                                     float* __restrict__ xio,
                                                     _Float16* __restrict__ x16, int N) {
    __shared__ _Float16 shB[64 * AGGW];     // 32 KB: 64 cols x 256 K, col-major
    int lane = threadIdx.x & 31;
    int wave = threadIdx.x >> 5;
    int rowBase = (blockIdx.x * 8 + wave) * 16;
    int colBase = blockIdx.y * 64;
    int half = lane >> 4;
    int l16  = lane & 15;

    if (threadIdx.x < 32) {
        tdm_load_1d_f16((unsigned)(uintptr_t)shB,
                        Wt + (size_t)colBase * AGGW, 64 * AGGW);
        __builtin_amdgcn_s_wait_tensorcnt(0);
    }
    __syncthreads();
    if (rowBase >= N) return;

    v8f acc[4] = {};
    const _Float16* arow = Agg + (size_t)(rowBase + l16) * AGGW + half * 16;
#pragma unroll
    for (int kc = 0; kc < AGGW; kc += 32) {
        v8h a0 = *(const v8h*)(arow + kc);
        v8h a1 = *(const v8h*)(arow + kc + 8);
        v16h A;
#pragma unroll
        for (int i = 0; i < 8; ++i) { A[i] = a0[i]; A[i + 8] = a1[i]; }
#pragma unroll
        for (int t = 0; t < 4; ++t) {
            const _Float16* bp = shB + (t * 16 + l16) * AGGW + kc + half * 16;
            v16h B = *(const v16h*)bp;
            acc[t] = __builtin_amdgcn_wmma_f32_16x16x32_f16(false, A, false, B,
                                                            (short)0, acc[t], false, false);
        }
    }
#pragma unroll
    for (int t = 0; t < 4; ++t)
#pragma unroll
        for (int g = 0; g < 8; ++g) {
            int row = rowBase + g + half * 8;
            int col = colBase + t * 16 + l16;
            float h = acc[t][g] + bo[col];
            float y = xio[(size_t)row * NN_H + col] + gelu_tanh(h);
            xio[(size_t)row * NN_H + col] = y;
            x16[(size_t)row * NN_H + col] = (_Float16)y;
        }
}

// ---------------------------------------------------------------------------
// Host launch
// ---------------------------------------------------------------------------
extern "C" void kernel_launch(void* const* d_in, const int* in_sizes, int n_in,
                              void* d_out, int out_size, void* d_ws, size_t ws_size,
                              hipStream_t stream) {
    const float* coords = (const float*)d_in[0];
    const float* fields = (const float*)d_in[1];
    const int*   ip0    = (const int*)d_in[2];
    const int*   ix0    = (const int*)d_in[3];
    const int*   ip1    = (const int*)d_in[4];
    const int*   ix1    = (const int*)d_in[5];
    const float* W_in   = (const float*)d_in[6];
    const float* b_in   = (const float*)d_in[7];
    const float* Wq     = (const float*)d_in[8];
    const float* Wk     = (const float*)d_in[9];
    const float* Wv     = (const float*)d_in[10];
    const float* Wo     = (const float*)d_in[11];
    const float* bo     = (const float*)d_in[12];

    const int N = in_sizes[0] / 2;   // coords is [N,2]
    float* x32 = (float*)d_out;      // x lives in d_out; updated in place per layer

    // Workspace carving (256B aligned)
    size_t off = 0;
    auto carve = [&](size_t bytes) {
        void* p = (char*)d_ws + off;
        off += (bytes + 255) & ~(size_t)255;
        return p;
    };
    _Float16* x16  = (_Float16*)carve((size_t)N * NN_H * 2);
    _Float16* qkv  = (_Float16*)carve((size_t)N * QKVW * 2);
    _Float16* agg  = (_Float16*)carve((size_t)N * AGGW * 2);
    float*    geom = (float*)carve((size_t)N * 6 * 4);
    _Float16* wqkv = (_Float16*)carve((size_t)NN_L * QKVW * NN_H * 2);
    _Float16* wot  = (_Float16*)carve((size_t)NN_L * NN_H * AGGW * 2);
    (void)ws_size; (void)n_in; (void)out_size;

    // 1) geometry
    geom_kernel<<<(N + 255) / 256, 256, 0, stream>>>(coords, ip0, ix0, ip1, ix1, geom, N);
    // 2) pack weights
    {
        int total = NN_L * QKVW * NN_H + NN_L * NN_H * AGGW;
        pack_weights<<<(total + 255) / 256, 256, 0, stream>>>(Wq, Wk, Wv, Wo, wqkv, wot);
    }
    // 3) prologue
    prologue_kernel<<<N, 128, 0, stream>>>(fields, geom, W_in, b_in, x32, x16, N);

    const int rowTiles  = (N + 15) / 16;
    const int rowBlocks = (rowTiles + 7) / 8;

    for (int l = 0; l < NN_L; ++l) {
        // QKV projections for both scales: [N,128] x [128,512]
        wmma_gemm_qkv<<<dim3(rowBlocks, QKVW / 64), 256, 0, stream>>>(
            x16, wqkv + (size_t)l * QKVW * NN_H, qkv, N);
        // cosine normalization of q and k
        normalize_qk<<<(N * 4 + 7) / 8, 256, 0, stream>>>(qkv, N);
        // attention + aggregation per scale
        edge_attn<<<(N + 7) / 8, 256, 0, stream>>>(qkv, ip0, ix0, agg, N, 0, 64, 128, 0);
        edge_attn<<<(N + 7) / 8, 256, 0, stream>>>(qkv, ip1, ix1, agg, N, 256, 320, 384, 128);
        // output projection + gelu + residual
        wmma_gemm_out<<<dim3(rowBlocks, NN_H / 64), 256, 0, stream>>>(
            agg, wot + (size_t)l * NN_H * AGGW, bo + (size_t)l * NN_H, x32, x16, N);
    }
}
